// MV_LSTM_84593675862184
// MI455X (gfx1250) — compile-verified, matrix operand order checked by reading, not run on hardware
//
#include <hip/hip_runtime.h>

// Problem dimensions (fixed by the reference)
#define B_SZ 1024
#define T_SZ 512
#define F_SZ 64
#define H_SZ 20
#define G_SZ 80
#define NTILES 5            // G_SZ / 16
#define BTILES (B_SZ / 16)  // 64

typedef __attribute__((ext_vector_type(16))) __bf16 v16bf;
typedef __attribute__((ext_vector_type(8)))  float  v8f;
typedef __attribute__((ext_vector_type(4)))  float  v4f;
typedef __attribute__((ext_vector_type(2)))  float  v2f;

__device__ __forceinline__ unsigned short f32_to_bf16_bits(float f) {
  union { float f; unsigned u; } v; v.f = f;
  unsigned r = v.u + 0x7FFFu + ((v.u >> 16) & 1u);   // round-to-nearest-even
  return (unsigned short)(r >> 16);
}
__device__ __forceinline__ __bf16 f32_to_bf16(float f) {
  unsigned short h = f32_to_bf16_bits(f);
  return __builtin_bit_cast(__bf16, h);
}

// Branchless activations: v_exp_f32 + v_rcp_f32 only (no IEEE div sequence,
// no divergent piecewise paths).
__device__ __forceinline__ float fast_rcp(float x) {
  return __builtin_amdgcn_rcpf(x);
}
__device__ __forceinline__ float fast_sigmoid(float x) {
  return fast_rcp(1.f + __expf(-x));
}
__device__ __forceinline__ float fast_tanh(float x) {
  // tanh(x) = 1 - 2/(1+exp(2x)); saturates correctly as exp -> 0 or inf.
  return 1.f - 2.f * fast_rcp(1.f + __expf(2.f * x));
}

// ISA 7.12.2: 16-bit A (16x32) / B (32x16) fragment: element i of the v16bf
// held by `lane` corresponds to K = (i&7) + (i>=8 ? 16:0) + (lane>=16 ? 8:0).
__device__ __forceinline__ int frag_k(int lane, int i) {
  return (i & 7) + ((i >= 8) ? 16 : 0) + ((lane >= 16) ? 8 : 0);
}

// Gin is stored tiled: [t][btile][nt][N=16][M=16] f32, so a lane's 8
// accumulator elements (fixed N, M = mhi..mhi+7) are contiguous (2x b128).
__device__ __forceinline__ size_t gin_tile_base(int t, int bt, int nt) {
  return (((size_t)t * BTILES + bt) * NTILES + nt) * 256;
}

// ---------------------------------------------------------------------------
// Kernel A: parallel input transform  Gin = xin @ Wih.T + (bih+bhh)
// One wave per 16-row tile of the flattened (t,b) dimension; loops the 5
// gate N-tiles reusing the A fragments. LAYER0 reads x[B,T,64] (K=64, two
// k-steps); later layers read ys[T,B,20] (K=20 padded to 32, one k-step).
// ---------------------------------------------------------------------------
template<bool LAYER0>
__global__ void lstm_input_gemm(const float* __restrict__ xin,
                                const float* __restrict__ Wih,
                                const float* __restrict__ bih,
                                const float* __restrict__ bhh,
                                float* __restrict__ Gout) {
  const int lane = threadIdx.x;            // 32 threads / block
  const int t    = blockIdx.x >> 6;        // tile r0 = blockIdx*16 over t*B+b
  const int bt   = blockIdx.x & (BTILES - 1);
  const int b0   = bt * 16;
  const int m    = lane & 15;              // A row within tile
  const int n    = lane & 15;              // C/B column within tile
  const int mhi  = (lane >= 16) ? 8 : 0;   // C row offset
  const int Fin  = LAYER0 ? F_SZ : H_SZ;

  v16bf afrag[2];
  #pragma unroll
  for (int ks = 0; ks < (LAYER0 ? 2 : 1); ++ks) {
    #pragma unroll
    for (int i = 0; i < 16; ++i) {
      int k = ks * 32 + frag_k(lane, i);
      float v = 0.f;
      if (k < Fin) {
        if (LAYER0) v = xin[((size_t)(b0 + m) * T_SZ + t) * F_SZ + k];
        else        v = xin[((size_t)t * B_SZ + (b0 + m)) * H_SZ + k];
      }
      afrag[ks][i] = f32_to_bf16(v);
    }
  }

  for (int nt = 0; nt < NTILES; ++nt) {
    const int gcol = nt * 16 + n;
    const float bias = bih[gcol] + bhh[gcol];
    v8f acc;
    #pragma unroll
    for (int j = 0; j < 8; ++j) acc[j] = bias;   // C preloaded with bias

    #pragma unroll
    for (int ks = 0; ks < (LAYER0 ? 2 : 1); ++ks) {
      v16bf bfrag;
      #pragma unroll
      for (int i = 0; i < 16; ++i) {
        int k = ks * 32 + frag_k(lane, i);
        bfrag[i] = f32_to_bf16((k < Fin) ? Wih[gcol * Fin + k] : 0.f);
      }
      acc = __builtin_amdgcn_wmma_f32_16x16x32_bf16(
          false, afrag[ks], false, bfrag, (short)0, acc, false, false);
    }
    // Tiled store: contiguous per lane -> 2x global_store_b128.
    float* gp = Gout + gin_tile_base(t, bt, nt) + n * 16 + mhi;
    *(v4f*)(gp)     = v4f{acc[0], acc[1], acc[2], acc[3]};
    *(v4f*)(gp + 4) = v4f{acc[4], acc[5], acc[6], acc[7]};
  }
}

// ---------------------------------------------------------------------------
// Kernel B: sequential recurrence. 64 blocks x 160 threads (5 waves).
// Each block owns a 16-row batch tile; wave w owns gate N-tile w.
// Per step: gates = Gin_tile (f32, L2-resident) + h @ Whh.T via one
// v_wmma_f32_16x16x32_bf16 (K=20 padded to 32), then the LSTM cell
// elementwise through a 16x80 LDS staging tile. h kept in LDS as bf16 in
// A-fragment layout; c kept in LDS as f32.
// ---------------------------------------------------------------------------
__global__ void lstm_recurrence(const float* __restrict__ Gin,
                                const float* __restrict__ Whh,
                                float* __restrict__ ys) {
  __shared__ unsigned short h_bf[16 * 32];   // [row][k], K padded to 32 (zeros)
  __shared__ float          c_st[16 * H_SZ];
  __shared__ float          gates[16 * G_SZ];

  const int tid  = threadIdx.x;          // 0..159
  const int lane = tid & 31;
  const int wave = tid >> 5;             // gate N-tile index 0..4
  const int bt   = blockIdx.x;
  const int b0   = bt * 16;
  const int m    = lane & 15;
  const int n    = lane & 15;
  const int gcol = wave * 16 + n;
  const int mhi  = (lane >= 16) ? 8 : 0;

  for (int i = tid; i < 16 * 32;   i += blockDim.x) h_bf[i] = 0;
  for (int i = tid; i < 16 * H_SZ; i += blockDim.x) c_st[i] = 0.f;
  __syncthreads();

  // Loop-invariant B fragment: Whh[gcol][k], k<20 else 0.
  v16bf whh;
  #pragma unroll
  for (int i = 0; i < 16; ++i) {
    int k = frag_k(lane, i);
    whh[i] = f32_to_bf16((k < H_SZ) ? Whh[gcol * H_SZ + k] : 0.f);
  }

  // Each thread owns cells (2*tid, 2*tid+1): same row always (rows are 20
  // wide, pairs start even), so all per-thread accesses merge to 64-bit ops.
  const int cellb = tid * 2;
  const int row   = cellb / H_SZ;
  const int c0    = cellb - row * H_SZ;
  const float* grow = gates + row * G_SZ + c0;
  unsigned*    hrow = (unsigned*)&h_bf[row * 32 + c0];   // c0 even -> aligned

  for (int t = 0; t < T_SZ; ++t) {
    // A fragment from LDS-resident h (bf16): 2x ds_load_b128.
    v16bf afrag;
    #pragma unroll
    for (int i = 0; i < 16; ++i) {
      unsigned short hb = h_bf[m * 32 + frag_k(lane, i)];
      afrag[i] = __builtin_bit_cast(__bf16, hb);
    }
    // C = precomputed input gates; tiled layout -> 2x global_load_b128.
    const float* gp = Gin + gin_tile_base(t, bt, wave) + n * 16 + mhi;
    v4f clo = *(const v4f*)(gp);
    v4f chi = *(const v4f*)(gp + 4);
    v8f acc = {clo[0], clo[1], clo[2], clo[3], chi[0], chi[1], chi[2], chi[3]};

    acc = __builtin_amdgcn_wmma_f32_16x16x32_bf16(
        false, afrag, false, whh, (short)0, acc, false, false);

    #pragma unroll
    for (int j = 0; j < 8; ++j) gates[(mhi + j) * G_SZ + gcol] = acc[j];
    __syncthreads();

    // Elementwise cell update: 320 cells, 2 adjacent cells per thread.
    v2f gi = *(const v2f*)(grow);
    v2f gf = *(const v2f*)(grow + 20);
    v2f gg = *(const v2f*)(grow + 40);
    v2f go = *(const v2f*)(grow + 60);
    v2f cprev = *(const v2f*)(c_st + cellb);

    float ca = fast_sigmoid(gf[0]) * cprev[0] + fast_sigmoid(gi[0]) * fast_tanh(gg[0]);
    float cb = fast_sigmoid(gf[1]) * cprev[1] + fast_sigmoid(gi[1]) * fast_tanh(gg[1]);
    float ha = fast_sigmoid(go[0]) * fast_tanh(ca);
    float hb = fast_sigmoid(go[1]) * fast_tanh(cb);

    *(v2f*)(c_st + cellb) = v2f{ca, cb};
    *hrow = (unsigned)f32_to_bf16_bits(ha) |
            ((unsigned)f32_to_bf16_bits(hb) << 16);
    float* yrow = ys + ((size_t)t * B_SZ + b0) * H_SZ;
    *(v2f*)(yrow + cellb) = v2f{ha, hb};    // one global_store_b64, coalesced

    // Prefetch next step's 5KB gate tile (160 threads x 32B stride, no
    // divergence; locality=3 -> near-cache prefetch).
    if (t + 1 < T_SZ)
      __builtin_prefetch(Gin + gin_tile_base(t + 1, bt, 0) + tid * 8, 0, 3);
    __syncthreads();
  }
}

// ---------------------------------------------------------------------------
// Final projection: out[b,o] = sum_j ys_flat[b,j] * Wlin[o,j] + blin[o].
// Pure streaming reduction (40 MB), memory-bound -> plain VALU.
// ---------------------------------------------------------------------------
__global__ void final_linear(const float* __restrict__ ys,
                             const float* __restrict__ Wlin,
                             const float* __restrict__ blin,
                             float* __restrict__ out) {
  __shared__ float red0[256];
  __shared__ float red1[256];
  const int b   = blockIdx.x;
  const int tid = threadIdx.x;
  float s0 = 0.f, s1 = 0.f;
  for (int j = tid; j < T_SZ * H_SZ; j += 256) {
    int t  = j / H_SZ;
    int hc = j - t * H_SZ;
    float v = ys[((size_t)t * B_SZ + b) * H_SZ + hc];
    s0 += v * Wlin[j];
    s1 += v * Wlin[T_SZ * H_SZ + j];
  }
  red0[tid] = s0; red1[tid] = s1;
  __syncthreads();
  for (int off = 128; off > 0; off >>= 1) {
    if (tid < off) { red0[tid] += red0[tid + off]; red1[tid] += red1[tid + off]; }
    __syncthreads();
  }
  if (tid == 0) {
    out[b * 2 + 0] = red0[0] + blin[0];
    out[b * 2 + 1] = red1[0] + blin[1];
  }
}

extern "C" void kernel_launch(void* const* d_in, const int* in_sizes, int n_in,
                              void* d_out, int out_size, void* d_ws, size_t ws_size,
                              hipStream_t stream) {
  const float* x    = (const float*)d_in[0];
  const float* Wih0 = (const float*)d_in[1];
  const float* Whh0 = (const float*)d_in[2];
  const float* bih0 = (const float*)d_in[3];
  const float* bhh0 = (const float*)d_in[4];
  const float* Wih1 = (const float*)d_in[5];
  const float* Whh1 = (const float*)d_in[6];
  const float* bih1 = (const float*)d_in[7];
  const float* bhh1 = (const float*)d_in[8];
  const float* Wih2 = (const float*)d_in[9];
  const float* Whh2 = (const float*)d_in[10];
  const float* bih2 = (const float*)d_in[11];
  const float* bhh2 = (const float*)d_in[12];
  const float* Wlin = (const float*)d_in[13];
  const float* blin = (const float*)d_in[14];
  float* out = (float*)d_out;

  // Workspace: Gin (T*B*80 f32, 167 MB, L2-resident, tiled layout) + two
  // h ping-pong buffers (T*B*20 f32, 40 MB each).
  float* Gin = (float*)d_ws;
  const size_t g_elems = (size_t)T_SZ * B_SZ * G_SZ;
  float* ysA = Gin + g_elems;
  const size_t y_elems = (size_t)T_SZ * B_SZ * H_SZ;
  float* ysB = ysA + y_elems;

  const dim3 gA((T_SZ * B_SZ) / 16), bA(32);
  const dim3 gB(BTILES, 1, 1), bB(160);

  lstm_input_gemm<true ><<<gA, bA, 0, stream>>>(x,   Wih0, bih0, bhh0, Gin);
  lstm_recurrence       <<<gB, bB, 0, stream>>>(Gin, Whh0, ysA);
  lstm_input_gemm<false><<<gA, bA, 0, stream>>>(ysA, Wih1, bih1, bhh1, Gin);
  lstm_recurrence       <<<gB, bB, 0, stream>>>(Gin, Whh1, ysB);
  lstm_input_gemm<false><<<gA, bA, 0, stream>>>(ysB, Wih2, bih2, bhh2, Gin);
  lstm_recurrence       <<<gB, bB, 0, stream>>>(Gin, Whh2, ysA);
  final_linear          <<<dim3(B_SZ), dim3(256), 0, stream>>>(ysA, Wlin, blin, out);
}
